// PartitionAttentionCl_644245094975
// MI455X (gfx1250) — matrix operand order repaired and checked by reading, not compiled, not run on hardware
//
#include <hip/hip_runtime.h>
#include <hip/hip_bf16.h>

// ---------- types for CDNA5 WMMA (wave32, 16x16x32 bf16 -> f32) ----------
typedef __bf16 bf16_t;
typedef bf16_t v16bf __attribute__((ext_vector_type(16)));
typedef float  v8f   __attribute__((ext_vector_type(8)));

union Frag {
    v16bf v;
    unsigned short us[16];
    unsigned int   u32[8];
    uint4          u128[2];
};

// gfx1250 async global->LDS path (ASYNCcnt) if the toolchain exposes it
#if defined(__has_builtin)
#if __has_builtin(__builtin_amdgcn_global_load_async_to_lds_b128) && \
    __has_builtin(__builtin_amdgcn_s_wait_asynccnt)
#define HAVE_ASYNC_LDS 1
#endif
#endif
#ifndef HAVE_ASYNC_LDS
#define HAVE_ASYNC_LDS 0
#endif

#if HAVE_ASYNC_LDS
// builtin signature (from compiler diagnostic): param0 = AS1 pointer to
// 128-bit int vector, param1 = AS3 (LDS) pointer, then imm offset, imm cpol.
typedef int b128_t __attribute__((vector_size(16)));
typedef __attribute__((address_space(1))) b128_t* glb_b128_p;
typedef __attribute__((address_space(3))) b128_t* lds_b128_p;
#endif

__device__ __forceinline__ unsigned short f2bf(float f) {
    unsigned int u = __float_as_uint(f);
    u += 0x7fffu + ((u >> 16) & 1u);          // round-to-nearest-even
    return (unsigned short)(u >> 16);
}

__device__ __forceinline__ v8f v8f_zero() {
    v8f z;
#pragma unroll
    for (int i = 0; i < 8; ++i) z[i] = 0.0f;
    return z;
}

__device__ __forceinline__ v8f wmma_bf16(const Frag& a, const Frag& b, v8f c) {
    return __builtin_amdgcn_wmma_f32_16x16x32_bf16(false, a.v, false, b.v,
                                                   (short)0, c, false, false);
}

// ---------------------- f32 -> bf16 weight convert -----------------------
__global__ void __launch_bounds__(256)
cvt_f32_bf16(const float* __restrict__ in, unsigned short* __restrict__ out, int n) {
    int i = blockIdx.x * 256 + threadIdx.x;
    if (i < n) out[i] = f2bf(in[i]);
}

// ------------- LayerNorm (+ optional window partition) -> bf16 -----------
// one block (128 threads = 4 waves) per token; C = 512
__global__ void __launch_bounds__(128)
ln_kernel(const float* __restrict__ x, const float* __restrict__ g,
          const float* __restrict__ b, unsigned short* __restrict__ y,
          int windowed) {
    const int tok = blockIdx.x;
    const int tid = threadIdx.x;
    const float4 v = ((const float4*)(x + (size_t)tok * 512))[tid];
    float s  = v.x + v.y + v.z + v.w;
    float sq = v.x * v.x + v.y * v.y + v.z * v.z + v.w * v.w;
#pragma unroll
    for (int d = 1; d < 32; d <<= 1) {
        s  += __shfl_xor(s,  d, 32);
        sq += __shfl_xor(sq, d, 32);
    }
    __shared__ float red[8];
    const int wv = tid >> 5;
    if ((tid & 31) == 0) { red[wv * 2] = s; red[wv * 2 + 1] = sq; }
    __syncthreads();
    s  = red[0] + red[2] + red[4] + red[6];
    sq = red[1] + red[3] + red[5] + red[7];
    const float mu   = s * (1.0f / 512.0f);
    const float var  = sq * (1.0f / 512.0f) - mu * mu;
    const float rstd = rsqrtf(var + 1e-6f);

    size_t orow;
    if (windowed) {
        int bimg = tok / 9216, rem = tok % 9216;
        int hh = rem / 96, ww = rem % 96;
        orow = (size_t)(((bimg * 12 + (hh >> 3)) * 12 + (ww >> 3)) * 64 +
                        (hh & 7) * 8 + (ww & 7));
    } else {
        orow = (size_t)tok;
    }
    const float4 gg = ((const float4*)g)[tid];
    const float4 bb = ((const float4*)b)[tid];
    unsigned int p0 = ((unsigned int)f2bf((v.y - mu) * rstd * gg.y + bb.y) << 16) |
                      (unsigned int)f2bf((v.x - mu) * rstd * gg.x + bb.x);
    unsigned int p1 = ((unsigned int)f2bf((v.w - mu) * rstd * gg.w + bb.w) << 16) |
                      (unsigned int)f2bf((v.z - mu) * rstd * gg.z + bb.z);
    uint2 pk; pk.x = p0; pk.y = p1;
    ((uint2*)(y + orow * 512))[tid] = pk;
}

// -------------------------- WMMA bf16 GEMM -------------------------------
// C[M x Nout] = A[M x K] * B[K x Nout] (+ epilogue).  M = 73728, tiles:
// block 128x64, 8 waves (4x2), wave 32x32.  EPI: 0 qkv, 1 proj, 2 glu, 3 fc2
template <int EPI>
__global__ void __launch_bounds__(256)
gemm_bf16_kernel(const unsigned short* __restrict__ A,
                 const unsigned short* __restrict__ B,
                 const unsigned short* __restrict__ B2,
                 const float* __restrict__ bias,
                 const float* __restrict__ bias2,
                 const float* __restrict__ ls,
                 const float* __restrict__ resid,
                 float* __restrict__ outf,
                 unsigned short* __restrict__ outb,
                 int K, int Nout, int ldB) {
    __shared__ __align__(16) unsigned short sA[128 * 32];
    __shared__ __align__(16) unsigned short sB[64 * 32];
    __shared__ __align__(16) unsigned short sB2[(EPI == 2) ? 64 * 32 : 8];

    const int tid  = threadIdx.x;
    const int lane = tid & 31;
    const int wave = tid >> 5;
    const int wrow = wave & 3;   // 4 row-waves * 32
    const int wcol = wave >> 2;  // 2 col-waves * 32
    const int r    = lane & 15;
    const int h    = lane >> 4;

    const int rowbase = blockIdx.y * 128;
    const int colbase = blockIdx.x * 64;

    const int a_row = tid >> 1;
    const int a_kc  = (tid & 1) * 16;
    const int b_k   = tid >> 3;
    const int b_n0  = (tid & 7) * 8;

    v8f acc[2][2], acc2[2][2];
#pragma unroll
    for (int i = 0; i < 2; ++i)
#pragma unroll
        for (int j = 0; j < 2; ++j) {
            acc[i][j] = v8f_zero();
            if (EPI == 2) acc2[i][j] = v8f_zero();
        }

    for (int k0 = 0; k0 < K; k0 += 32) {
        // ---- stage A tile 128x32 (row-major in LDS) ----
        const unsigned short* gA = A + (size_t)(rowbase + a_row) * K + k0 + a_kc;
        unsigned short* lA = sA + a_row * 32 + a_kc;
#if HAVE_ASYNC_LDS
        // CDNA5 async DMA: global -> LDS without VGPR transit (ASYNCcnt)
        __builtin_amdgcn_global_load_async_to_lds_b128(
            (glb_b128_p)(gA), (lds_b128_p)(lA), 0, 0);
        __builtin_amdgcn_global_load_async_to_lds_b128(
            (glb_b128_p)(gA + 8), (lds_b128_p)(lA + 8), 0, 0);
#else
        {
            const uint4* pa = (const uint4*)gA;
            uint4 v0 = pa[0];
            uint4 v1 = pa[1];
            *(uint4*)(lA)     = v0;
            *(uint4*)(lA + 8) = v1;
        }
#endif
        // ---- stage B tile 32x64 -> column-major sB[n*32 + k] ----
        {
            union { uint4 q; unsigned short s[8]; } t;
            t.q = *(const uint4*)(B + (size_t)(k0 + b_k) * ldB + colbase + b_n0);
#pragma unroll
            for (int j = 0; j < 8; ++j) sB[(b_n0 + j) * 32 + b_k] = t.s[j];
            if (EPI == 2) {
                t.q = *(const uint4*)(B2 + (size_t)(k0 + b_k) * ldB + colbase + b_n0);
#pragma unroll
                for (int j = 0; j < 8; ++j) sB2[(b_n0 + j) * 32 + b_k] = t.s[j];
            }
        }
        if (k0 + 32 < K) {  // gfx1250 global_prefetch of next K-tile
            __builtin_prefetch(A + (size_t)(rowbase + a_row) * K + k0 + 32 + a_kc, 0, 1);
            __builtin_prefetch(B + (size_t)(k0 + 32 + b_k) * ldB + colbase + b_n0, 0, 1);
        }
#if HAVE_ASYNC_LDS
        __builtin_amdgcn_s_wait_asynccnt(0);
#endif
        __syncthreads();

        Frag a[2], b[2], b2[2];
#pragma unroll
        for (int mi = 0; mi < 2; ++mi) {
            const unsigned short* p = sA + (wrow * 32 + mi * 16 + r) * 32 + 16 * h;
            a[mi].u128[0] = *(const uint4*)(p);
            a[mi].u128[1] = *(const uint4*)(p + 8);
        }
#pragma unroll
        for (int ni = 0; ni < 2; ++ni) {
            const unsigned short* p = sB + (wcol * 32 + ni * 16 + r) * 32 + 16 * h;
            b[ni].u128[0] = *(const uint4*)(p);
            b[ni].u128[1] = *(const uint4*)(p + 8);
            if (EPI == 2) {
                const unsigned short* p2 = sB2 + (wcol * 32 + ni * 16 + r) * 32 + 16 * h;
                b2[ni].u128[0] = *(const uint4*)(p2);
                b2[ni].u128[1] = *(const uint4*)(p2 + 8);
            }
        }
#pragma unroll
        for (int mi = 0; mi < 2; ++mi)
#pragma unroll
            for (int ni = 0; ni < 2; ++ni) {
                acc[mi][ni] = wmma_bf16(a[mi], b[ni], acc[mi][ni]);
                if (EPI == 2) acc2[mi][ni] = wmma_bf16(a[mi], b2[ni], acc2[mi][ni]);
            }
        __syncthreads();
    }

    // ------------------------------ epilogue -----------------------------
#pragma unroll
    for (int mi = 0; mi < 2; ++mi)
#pragma unroll
        for (int ni = 0; ni < 2; ++ni)
#pragma unroll
            for (int e = 0; e < 8; ++e) {
                const int gm = rowbase + wrow * 32 + mi * 16 + e + 8 * h;
                const int gn = colbase + wcol * 32 + ni * 16 + r;
                float val = acc[mi][ni][e];
                if (EPI == 0) {                 // qkv: +bias -> bf16
                    val += bias[gn];
                    outb[(size_t)gm * Nout + gn] = f2bf(val);
                } else if (EPI == 1) {          // proj: window-reverse + ls1 resid
                    val += bias[gn];
                    int win = gm >> 6, t = gm & 63;
                    int bi = win / 144, rem = win % 144;
                    int wh = rem / 12, ww = rem % 12;
                    int sp = (bi * 96 + wh * 8 + (t >> 3)) * 96 + ww * 8 + (t & 7);
                    size_t o = (size_t)sp * 512 + gn;
                    outf[o] = resid[o] + ls[gn] * val;
                } else if (EPI == 2) {          // glu: h * silu(gate) -> bf16
                    float hv = val + bias[gn];
                    float gv = acc2[mi][ni][e] + bias2[gn];
                    float sg = gv / (1.0f + __expf(-gv));
                    outb[(size_t)gm * Nout + gn] = f2bf(hv * sg);
                } else {                        // fc2: + ls2 resid -> f32 out
                    val += bias[gn];
                    size_t o = (size_t)gm * Nout + gn;
                    outf[o] = resid[o] + ls[gn] * val;
                }
            }
}

// ----------------------- windowed multi-head attention -------------------
// one block per window (1152), 4 waves, each wave does 4 heads serially.
// qkv rows are windowed tokens, cols: q[h*32+d], k[512+...], v[1024+...]
__global__ void __launch_bounds__(128)
attn_kernel(const unsigned short* __restrict__ qkv,
            unsigned short* __restrict__ obuf) {
    __shared__ __align__(16) unsigned short sP[4][64 * 64];
    const int win  = blockIdx.x;
    const int lane = threadIdx.x & 31;
    const int wave = threadIdx.x >> 5;
    const int r = lane & 15, h = lane >> 4;
    const unsigned short* base = qkv + (size_t)win * 64 * 1536;
    unsigned short* P = sP[wave];
    const float scale = 0.17677669529663687f;  // 1/sqrt(32)

    for (int hi = 0; hi < 4; ++hi) {
        const int head = wave + hi * 4;
        const int qc   = head * 32;

        // ---- Q A-fragments (16x32 each, 4 M-tiles) straight from global
        Frag qa[4];
#pragma unroll
        for (int mt = 0; mt < 4; ++mt) {
            const unsigned short* p = base + (size_t)(mt * 16 + r) * 1536 + qc + 16 * h;
            qa[mt].u128[0] = *(const uint4*)(p);
            qa[mt].u128[1] = *(const uint4*)(p + 8);
        }
        // ---- K^T B-fragments (32x16): lane n = token, contiguous dims
        Frag kb[4];
#pragma unroll
        for (int nt = 0; nt < 4; ++nt) {
            const unsigned short* p = base + (size_t)(nt * 16 + r) * 1536 + 512 + qc + 16 * h;
            kb[nt].u128[0] = *(const uint4*)(p);
            kb[nt].u128[1] = *(const uint4*)(p + 8);
        }
        // ---- S = Q K^T  (4x4 tiles, one wmma each since K=32)
        v8f s[4][4];
#pragma unroll
        for (int mt = 0; mt < 4; ++mt)
#pragma unroll
            for (int nt = 0; nt < 4; ++nt) {
                s[mt][nt] = v8f_zero();
                s[mt][nt] = wmma_bf16(qa[mt], kb[nt], s[mt][nt]);
            }
        // ---- softmax rows (each lane-half owns disjoint M rows -> width-16)
#pragma unroll
        for (int mt = 0; mt < 4; ++mt)
#pragma unroll
            for (int e = 0; e < 8; ++e) {
                float mx = s[mt][0][e];
#pragma unroll
                for (int nt = 1; nt < 4; ++nt) mx = fmaxf(mx, s[mt][nt][e]);
#pragma unroll
                for (int d = 1; d < 16; d <<= 1) mx = fmaxf(mx, __shfl_xor(mx, d, 16));
                float sum = 0.0f;
#pragma unroll
                for (int nt = 0; nt < 4; ++nt) {
                    float t = __expf((s[mt][nt][e] - mx) * scale);
                    s[mt][nt][e] = t;
                    sum += t;
                }
#pragma unroll
                for (int d = 1; d < 16; d <<= 1) sum += __shfl_xor(sum, d, 16);
                const float rs = 1.0f / sum;
#pragma unroll
                for (int nt = 0; nt < 4; ++nt) s[mt][nt][e] *= rs;
            }
        // ---- stash P (bf16) in wave-private LDS, row-major 64x64
#pragma unroll
        for (int mt = 0; mt < 4; ++mt)
#pragma unroll
            for (int nt = 0; nt < 4; ++nt)
#pragma unroll
                for (int e = 0; e < 8; ++e)
                    P[(mt * 16 + e + 8 * h) * 64 + nt * 16 + r] = f2bf(s[mt][nt][e]);
        // ---- V B-fragments (64x32, 2 k-blocks x 2 n-tiles)
        Frag vb[2][2];
#pragma unroll
        for (int kk2 = 0; kk2 < 2; ++kk2)
#pragma unroll
            for (int nt2 = 0; nt2 < 2; ++nt2) {
#pragma unroll
                for (int j = 0; j < 8; ++j) {
                    const int kk = kk2 * 32 + 16 * h + 2 * j;
                    const size_t c = 1024 + (size_t)qc + nt2 * 16 + r;
                    vb[kk2][nt2].us[2 * j]     = base[(size_t)kk * 1536 + c];
                    vb[kk2][nt2].us[2 * j + 1] = base[(size_t)(kk + 1) * 1536 + c];
                }
            }
        // ---- O = P V
        v8f o[4][2];
#pragma unroll
        for (int mt = 0; mt < 4; ++mt)
#pragma unroll
            for (int nt2 = 0; nt2 < 2; ++nt2) o[mt][nt2] = v8f_zero();
#pragma unroll
        for (int mt = 0; mt < 4; ++mt)
#pragma unroll
            for (int kk2 = 0; kk2 < 2; ++kk2) {
                Frag pa;
                const unsigned short* p = P + (mt * 16 + r) * 64 + kk2 * 32 + 16 * h;
                pa.u128[0] = *(const uint4*)(p);
                pa.u128[1] = *(const uint4*)(p + 8);
#pragma unroll
                for (int nt2 = 0; nt2 < 2; ++nt2)
                    o[mt][nt2] = wmma_bf16(pa, vb[kk2][nt2], o[mt][nt2]);
            }
        // ---- store O as bf16 (windowed rows, head columns)
#pragma unroll
        for (int mt = 0; mt < 4; ++mt)
#pragma unroll
            for (int nt2 = 0; nt2 < 2; ++nt2)
#pragma unroll
                for (int e = 0; e < 8; ++e) {
                    const int m = mt * 16 + e + 8 * h;
                    obuf[((size_t)win * 64 + m) * 512 + qc + nt2 * 16 + r] =
                        f2bf(o[mt][nt2][e]);
                }
    }
}

// ------------------------------- launcher --------------------------------
extern "C" void kernel_launch(void* const* d_in, const int* in_sizes, int n_in,
                              void* d_out, int out_size, void* d_ws, size_t ws_size,
                              hipStream_t stream) {
    const float* x      = (const float*)d_in[0];
    const float* n1g    = (const float*)d_in[1];
    const float* n1b    = (const float*)d_in[2];
    const float* qkv_w  = (const float*)d_in[3];
    const float* qkv_b  = (const float*)d_in[4];
    const float* proj_w = (const float*)d_in[5];
    const float* proj_b = (const float*)d_in[6];
    const float* ls1    = (const float*)d_in[7];
    const float* n2g    = (const float*)d_in[8];
    const float* n2b    = (const float*)d_in[9];
    const float* glu_w  = (const float*)d_in[10];
    const float* glu_b  = (const float*)d_in[11];
    const float* fc2_w  = (const float*)d_in[12];
    const float* fc2_b  = (const float*)d_in[13];
    const float* ls2    = (const float*)d_in[14];

    const size_t NT = 73728;  // 8*96*96 tokens
    char* ws = (char*)d_ws;
    size_t off = 0;
    auto take = [&](size_t bytes) -> char* {
        char* p = ws + off;
        off += (bytes + 255) & ~(size_t)255;
        return p;
    };
    unsigned short* y1   = (unsigned short*)take(NT * 512 * 2);    // LN1, windowed
    unsigned short* qkvb = (unsigned short*)take(NT * 1536 * 2);   // qkv, windowed
    unsigned short* ob   = (unsigned short*)take(NT * 512 * 2);    // attn out, windowed
    float*          x1   = (float*)take(NT * 512 * 4);             // residual 1, spatial
    unsigned short* y2   = (unsigned short*)take(NT * 512 * 2);    // LN2, spatial
    unsigned short* hb   = (unsigned short*)take(NT * 1344 * 2);   // glu out, spatial
    unsigned short* wq   = (unsigned short*)take(512 * 1536 * 2);
    unsigned short* wp   = (unsigned short*)take(512 * 512 * 2);
    unsigned short* wg   = (unsigned short*)take(512 * 2688 * 2);
    unsigned short* wf   = (unsigned short*)take(1344 * 512 * 2);

    cvt_f32_bf16<<<(512 * 1536 + 255) / 256, 256, 0, stream>>>(qkv_w, wq, 512 * 1536);
    cvt_f32_bf16<<<(512 * 512 + 255) / 256, 256, 0, stream>>>(proj_w, wp, 512 * 512);
    cvt_f32_bf16<<<(512 * 2688 + 255) / 256, 256, 0, stream>>>(glu_w, wg, 512 * 2688);
    cvt_f32_bf16<<<(1344 * 512 + 255) / 256, 256, 0, stream>>>(fc2_w, wf, 1344 * 512);

    // LN1 + window partition
    ln_kernel<<<(unsigned)NT, 128, 0, stream>>>(x, n1g, n1b, y1, 1);
    // QKV GEMM
    gemm_bf16_kernel<0><<<dim3(1536 / 64, 576), 256, 0, stream>>>(
        y1, wq, nullptr, qkv_b, nullptr, nullptr, nullptr, nullptr, qkvb, 512, 1536, 1536);
    // attention
    attn_kernel<<<1152, 128, 0, stream>>>(qkvb, ob);
    // proj GEMM + window-reverse + ls1 residual
    gemm_bf16_kernel<1><<<dim3(512 / 64, 576), 256, 0, stream>>>(
        ob, wp, nullptr, proj_b, nullptr, ls1, x, x1, nullptr, 512, 512, 512);
    // LN2
    ln_kernel<<<(unsigned)NT, 128, 0, stream>>>(x1, n2g, n2b, y2, 0);
    // fused SwiGLU GEMM (dual-B)
    gemm_bf16_kernel<2><<<dim3(1344 / 64, 576), 256, 0, stream>>>(
        y2, wg, wg + 1344, glu_b, glu_b + 1344, nullptr, nullptr, nullptr, hb, 512, 1344, 2688);
    // fc2 GEMM + ls2 residual -> d_out (f32)
    gemm_bf16_kernel<3><<<dim3(512 / 64, 576), 256, 0, stream>>>(
        hb, wf, nullptr, fc2_b, nullptr, ls2, x1, (float*)d_out, nullptr, 1344, 512, 512);
}
